// MyLinear_79319456023198
// MI455X (gfx1250) — matrix operand order repaired and checked by reading, not compiled, not run on hardware
//
#include <hip/hip_runtime.h>
#include <hip/hip_bf16.h>
#include <hip/hip_fp16.h>

// ---------------------------------------------------------------------------
// MI455X (gfx1250) implementation.
// Phase 1: sequential 16384-step scan, one workgroup, w pinned in 320KB LDS.
// Phase 2: fused GEMM(X @ w_final + bias) + row softmax using
//          v_wmma_f32_16x16x32_f16 (f32 accumulate), memory-bound at 23.3TB/s.
// ---------------------------------------------------------------------------

typedef __attribute__((ext_vector_type(16))) _Float16 v16h;
typedef __attribute__((ext_vector_type(8)))  float    v8f;

#define IN_U   256
#define OUT_U  256
#define HID    100
#define BATCH  65536
#define N_IDX  128
#define N_PAIR (N_IDX * N_IDX)
#define FEAT   (IN_U + OUT_U)   // 512
#define WPAD   257              // 257 % 64 == 1 -> conflict-free column reads

// ---------------------------------------------------------------------------
// Kernel 0: transpose W1 [512,100] -> W1T [100,512] and W2 [100,100] -> W2T,
// so the scan's per-hidden-unit dot products stream contiguous float4s.
// ---------------------------------------------------------------------------
__global__ void prep_transpose(const float* __restrict__ W1,
                               const float* __restrict__ W2,
                               float* __restrict__ W1T,
                               float* __restrict__ W2T) {
    int idx = blockIdx.x * blockDim.x + threadIdx.x;
    if (idx < FEAT * HID) {
        int t = idx / HID, k = idx % HID;
        W1T[k * FEAT + t] = W1[idx];
    }
    if (idx < HID * HID) {
        int t = idx / HID, k = idx % HID;
        W2T[k * HID + t] = W2[idx];
    }
}

// ---------------------------------------------------------------------------
// Kernel 1: the sequential scan. Single workgroup (256 threads = 8 wave32).
// The full 256x256 f32 weight matrix lives in LDS (CDNA5: 320KB per WGP).
// Epilogue writes w_final transposed as f16 for the WMMA GEMM.
// ---------------------------------------------------------------------------
__global__ __launch_bounds__(256) void scan_kernel(
    const float* __restrict__ weight,
    const float* __restrict__ b1, const float* __restrict__ b2,
    const float* __restrict__ W3, const float* __restrict__ b3,
    const int*   __restrict__ idx_in, const int* __restrict__ idx_out,
    const float* __restrict__ W1T, const float* __restrict__ W2T,
    _Float16*    __restrict__ wT16)
{
    extern __shared__ float smem[];
    float* w_s    = smem;                  // 256*257 floats
    float* feat_s = w_s + IN_U * WPAD;     // 512
    float* h1_s   = feat_s + FEAT;         // 128 (100 used, padded)
    float* h2_s   = h1_s + 128;            // 128
    const int tid = threadIdx.x;

    // Stage w into LDS with padded stride.
    for (int idx = tid; idx < IN_U * OUT_U; idx += 256)
        w_s[(idx >> 8) * WPAD + (idx & 255)] = weight[idx];
    __syncthreads();

    for (int p = 0; p < N_PAIR; ++p) {
        const int i = idx_in[p >> 7];    // i outer (repeat)
        const int j = idx_out[p & 127];  // j inner (tile)

        // feat = concat(w[i,:], w[:,j]) -- column read is bank-conflict-free.
        feat_s[tid]        = w_s[i * WPAD + tid];
        feat_s[IN_U + tid] = w_s[tid * WPAD + j];
        __syncthreads();

        // Layer 1: h1 = relu(feat @ W1 + b1), one hidden unit per thread.
        if (tid < HID) {
            const float4* wr = (const float4*)(W1T + tid * FEAT);
            const float4* fs = (const float4*)feat_s;
            float h = b1[tid];
            #pragma unroll 8
            for (int t = 0; t < FEAT / 4; ++t) {
                float4 a = fs[t], b = wr[t];
                h = fmaf(a.x, b.x, h); h = fmaf(a.y, b.y, h);
                h = fmaf(a.z, b.z, h); h = fmaf(a.w, b.w, h);
            }
            h1_s[tid] = fmaxf(h, 0.f);
        }
        __syncthreads();

        // Layer 2: h2 = relu(h1 @ W2 + b2)
        if (tid < HID) {
            const float4* wr = (const float4*)(W2T + tid * HID);
            const float4* fs = (const float4*)h1_s;
            float h = b2[tid];
            #pragma unroll 5
            for (int t = 0; t < HID / 4; ++t) {
                float4 a = fs[t], b = wr[t];
                h = fmaf(a.x, b.x, h); h = fmaf(a.y, b.y, h);
                h = fmaf(a.z, b.z, h); h = fmaf(a.w, b.w, h);
            }
            h2_s[tid] = fmaxf(h, 0.f);
        }
        __syncthreads();

        // Layer 3 + sigmoid - 0.5, wave0 reduction (wave32 shuffles).
        if (tid < 32) {
            float acc = 0.f;
            for (int t = tid; t < HID; t += 32) acc = fmaf(h2_s[t], W3[t], acc);
            for (int off = 16; off > 0; off >>= 1) acc += __shfl_down(acc, off);
            if (tid == 0) {
                float v = 1.f / (1.f + __expf(-(acc + b3[0]))) - 0.5f;
                w_s[i * WPAD + j] = v;
            }
        }
        __syncthreads();
    }

    // Epilogue: wT16[n][k] = (f16) w_final[k][n]  (B-side layout for WMMA).
    for (int idx = tid; idx < IN_U * OUT_U; idx += 256) {
        int n = idx >> 8, k = idx & 255;
        wT16[n * 256 + k] = (_Float16)w_s[k * WPAD + n];
    }
}

// ---------------------------------------------------------------------------
// Kernel 2: fused GEMM + softmax. Block = 16 rows x 256 cols, 8 wave32;
// each wave owns two 16x16 output tiles, K=256 in 8 steps of 32 via
// v_wmma_f32_16x16x32_f16. Softmax fused in-block (saves 134MB round trip).
// ---------------------------------------------------------------------------
__global__ __launch_bounds__(256) void gemm_softmax(
    const float*    __restrict__ X,
    const _Float16* __restrict__ wT16,
    const float*    __restrict__ bias,
    float*          __restrict__ out)
{
    __shared__ float logit_s[16][OUT_U];

    const int tid  = threadIdx.x;
    const int wave = tid >> 5;
    const int lane = tid & 31;
    const int l16  = lane & 15;
    const int hi   = lane >> 4;           // half-wave select
    const int m0   = blockIdx.x * 16;
    const int nt0  = wave * 2, nt1 = nt0 + 1;

    // A (16x32): lanes 0-15 hold K 0-15, lanes 16-31 hold K 16-31.
    // B (32x16): lanes 0-15 hold K 0-15,
    // lanes 16-31 hold K 16-31, contiguous per lane.
    const int kbA = hi ? 16 : 0;
    const int kbB = hi ? 16 : 0;

    const float*    xrow = X + (size_t)(m0 + l16) * IN_U;
    const _Float16* b0p  = wT16 + (size_t)(nt0 * 16 + l16) * 256;
    const _Float16* b1p  = wT16 + (size_t)(nt1 * 16 + l16) * 256;

    v8f c0 = {};
    v8f c1 = {};

    for (int k0 = 0; k0 < IN_U; k0 += 32) {
        if (k0 + 32 < IN_U)
            __builtin_prefetch(xrow + k0 + 32, 0, 1);  // global_prefetch_b8

        // A fragment: f32 loads, convert to f16 in registers (f32 accumulate).
        const float4* pa = (const float4*)(xrow + k0 + kbA);
        float4 a0 = pa[0], a1 = pa[1];
        const float4* pb = (const float4*)(xrow + k0 + kbA + 8);
        float4 a2 = pb[0], a3 = pb[1];
        v16h a;
        a[0]  = (_Float16)a0.x; a[1]  = (_Float16)a0.y;
        a[2]  = (_Float16)a0.z; a[3]  = (_Float16)a0.w;
        a[4]  = (_Float16)a1.x; a[5]  = (_Float16)a1.y;
        a[6]  = (_Float16)a1.z; a[7]  = (_Float16)a1.w;
        a[8]  = (_Float16)a2.x; a[9]  = (_Float16)a2.y;
        a[10] = (_Float16)a2.z; a[11] = (_Float16)a2.w;
        a[12] = (_Float16)a3.x; a[13] = (_Float16)a3.y;
        a[14] = (_Float16)a3.z; a[15] = (_Float16)a3.w;

        // B fragments: one contiguous 32-byte load per lane.
        v16h bb0 = *(const v16h*)(b0p + k0 + kbB);
        v16h bb1 = *(const v16h*)(b1p + k0 + kbB);

        c0 = __builtin_amdgcn_wmma_f32_16x16x32_f16(
                 false, a, false, bb0, (short)0, c0, false, false);
        c1 = __builtin_amdgcn_wmma_f32_16x16x32_f16(
                 false, a, false, bb1, (short)0, c1, false, false);
    }

    // C layout: VGPR r -> row r (lanes 0-15) / row 8+r (lanes 16-31), col=l16.
    const int col0 = nt0 * 16 + l16;
    const int col1 = nt1 * 16 + l16;
    #pragma unroll
    for (int r = 0; r < 8; ++r) {
        int row = r + hi * 8;
        logit_s[row][col0] = c0[r] + bias[col0];
        logit_s[row][col1] = c1[r] + bias[col1];
    }
    __syncthreads();

    // Fused softmax: tid = row*16 + sub; 16 lanes per row are consecutive
    // within a half-wave, so xor-shuffles 8/4/2/1 reduce each row.
    const int row = tid >> 4, sub = tid & 15;
    float mx = -3.402823466e38f;
    for (int c = sub; c < OUT_U; c += 16) mx = fmaxf(mx, logit_s[row][c]);
    for (int off = 8; off > 0; off >>= 1) mx = fmaxf(mx, __shfl_xor(mx, off));

    float sum = 0.f;
    for (int c = sub; c < OUT_U; c += 16) {
        float e = __expf(logit_s[row][c] - mx);
        logit_s[row][c] = e;   // own slots only, no race
        sum += e;
    }
    for (int off = 8; off > 0; off >>= 1) sum += __shfl_xor(sum, off);

    const float inv = 1.f / sum;
    float* orow = out + (size_t)(m0 + row) * OUT_U;
    for (int c = sub; c < OUT_U; c += 16) orow[c] = logit_s[row][c] * inv;
}

// ---------------------------------------------------------------------------
extern "C" void kernel_launch(void* const* d_in, const int* in_sizes, int n_in,
                              void* d_out, int out_size, void* d_ws, size_t ws_size,
                              hipStream_t stream) {
    const float* X      = (const float*)d_in[0];
    const float* weight = (const float*)d_in[1];
    const float* bias   = (const float*)d_in[2];
    const float* W1     = (const float*)d_in[3];
    const float* b1     = (const float*)d_in[4];
    const float* W2     = (const float*)d_in[5];
    const float* b2     = (const float*)d_in[6];
    const float* W3     = (const float*)d_in[7];
    const float* b3     = (const float*)d_in[8];
    const int* idx_in   = (const int*)d_in[9];
    const int* idx_out  = (const int*)d_in[10];
    float* out = (float*)d_out;

    // Workspace layout.
    char* ws = (char*)d_ws;
    _Float16* wT16 = (_Float16*)ws;                       // 131072 B
    float*    W1T  = (float*)(ws + 131072);               // 204800 B
    float*    W2T  = (float*)(ws + 131072 + 204800);      //  40000 B

    // Phase 0: weight transposes (L2-resident for the scan).
    prep_transpose<<<(FEAT * HID + 255) / 256, 256, 0, stream>>>(W1, W2, W1T, W2T);

    // Phase 1: sequential scan, one workgroup, big dynamic LDS (~260 KB).
    size_t smem = (size_t)(IN_U * WPAD + FEAT + 128 + 128) * sizeof(float);
    scan_kernel<<<1, 256, smem, stream>>>(weight, b1, b2, W3, b3,
                                          idx_in, idx_out, W1T, W2T, wT16);

    // Phase 2: WMMA GEMM + fused softmax. 4096 blocks x 16 rows.
    gemm_softmax<<<BATCH / 16, 256, 0, stream>>>(X, wT16, bias, out);
}
